// DeepPointNet2_62895501082991
// MI455X (gfx1250) — matrix operand order repaired and checked by reading, not compiled, not run on hardware
//
#include <hip/hip_runtime.h>
#include <hip/hip_bf16.h>

typedef __bf16 bf16;
typedef __attribute__((ext_vector_type(16))) __bf16 v16bf;
typedef __attribute__((ext_vector_type(8)))  __bf16 v8bf;
typedef __attribute__((ext_vector_type(8)))  float  v8f;

#define KNN 64
#define NPT 4096
#define NB  4

// ---------------------------------------------------------------------------
// Weight pad/convert: f32 [Ci,Co] -> bf16 [Cip,Co], zero rows Ci..Cip-1
// ---------------------------------------------------------------------------
__global__ void pad_weights_kernel(const float* __restrict__ W, bf16* __restrict__ Wp,
                                   int Ci, int Cip, int Co) {
    long long total = (long long)Cip * Co;
    long long stride = (long long)gridDim.x * blockDim.x;
    for (long long i = (long long)blockIdx.x * blockDim.x + threadIdx.x; i < total; i += stride) {
        int ci = (int)(i / Co), co = (int)(i % Co);
        float v = (ci < Ci) ? W[(size_t)ci * Co + co] : 0.0f;
        Wp[i] = (bf16)v;
    }
}

// ---------------------------------------------------------------------------
// Farthest point sampling: one block, sequential selection with LDS argmax.
// Tie-break to the lowest index (matches jnp.argmax first-occurrence).
// ---------------------------------------------------------------------------
__global__ void fps_kernel(const float* __restrict__ pos, int Nsrc, int M,
                           int* __restrict__ idx_out) {
    extern __shared__ float smem[];
    float* sd = smem;                 // [Nsrc]
    float* rv = sd + Nsrc;            // [blockDim]
    int*   ri = (int*)(rv + blockDim.x);
    int tid = threadIdx.x, nt = blockDim.x;

    float x0 = pos[0], y0 = pos[1], z0 = pos[2];
    for (int i = tid; i < Nsrc; i += nt) {
        float dx = pos[3*i] - x0, dy = pos[3*i+1] - y0, dz = pos[3*i+2] - z0;
        sd[i] = dx*dx + dy*dy + dz*dz;
    }
    if (tid == 0) idx_out[0] = 0;
    __syncthreads();

    for (int m = 1; m < M; ++m) {
        float bv = -1.0f; int bi = 0x7fffffff;
        for (int i = tid; i < Nsrc; i += nt) {
            float v = sd[i];
            if (v > bv || (v == bv && i < bi)) { bv = v; bi = i; }
        }
        rv[tid] = bv; ri[tid] = bi;
        __syncthreads();
        for (int s = nt >> 1; s > 0; s >>= 1) {
            if (tid < s) {
                float v2 = rv[tid + s]; int i2 = ri[tid + s];
                if (v2 > rv[tid] || (v2 == rv[tid] && i2 < ri[tid])) { rv[tid] = v2; ri[tid] = i2; }
            }
            __syncthreads();
        }
        int best = ri[0];
        if (tid == 0) idx_out[m] = best;
        float bx = pos[3*best], by = pos[3*best+1], bz = pos[3*best+2];
        __syncthreads();
        for (int i = tid; i < Nsrc; i += nt) {
            float dx = pos[3*i] - bx, dy = pos[3*i+1] - by, dz = pos[3*i+2] - bz;
            float nd = dx*dx + dy*dy + dz*dz;
            if (nd < sd[i]) sd[i] = nd;
        }
        __syncthreads();
    }
}

__global__ void gather_pos_kernel(const float* __restrict__ src, const int* __restrict__ idx,
                                  int M, float* __restrict__ dst) {
    int i = blockIdx.x * blockDim.x + threadIdx.x;
    if (i < M * 3) { int r = i / 3, c = i % 3; dst[i] = src[(size_t)idx[r] * 3 + c]; }
}

// ---------------------------------------------------------------------------
// Radius ball query: block per query; KNN rounds of argmin selection in LDS.
// Selection order is ascending distance (matches top_k of -d2).
// ---------------------------------------------------------------------------
__global__ void radius_topk_kernel(const float* __restrict__ pos_src, int Nsrc,
                                   const float* __restrict__ pos_q, float r2,
                                   int* __restrict__ nidx, unsigned char* __restrict__ nmask) {
    extern __shared__ float smem[];
    float* sd = smem;                 // [Nsrc]
    float* rv = sd + Nsrc;            // [blockDim]
    int*   ri = (int*)(rv + blockDim.x);
    int q = blockIdx.x, tid = threadIdx.x, nt = blockDim.x;

    float qx = pos_q[3*q], qy = pos_q[3*q+1], qz = pos_q[3*q+2];
    for (int i = tid; i < Nsrc; i += nt) {
        float dx = pos_src[3*i] - qx, dy = pos_src[3*i+1] - qy, dz = pos_src[3*i+2] - qz;
        sd[i] = dx*dx + dy*dy + dz*dz;
    }
    __syncthreads();
    for (int kk = 0; kk < KNN; ++kk) {
        float bv = 3.3e38f; int bi = 0x7fffffff;
        for (int i = tid; i < Nsrc; i += nt) {
            float v = sd[i];
            if (v < bv || (v == bv && i < bi)) { bv = v; bi = i; }
        }
        rv[tid] = bv; ri[tid] = bi;
        __syncthreads();
        for (int s = nt >> 1; s > 0; s >>= 1) {
            if (tid < s) {
                float v2 = rv[tid + s]; int i2 = ri[tid + s];
                if (v2 < rv[tid] || (v2 == rv[tid] && i2 < ri[tid])) { rv[tid] = v2; ri[tid] = i2; }
            }
            __syncthreads();
        }
        if (tid == 0) {
            int best = ri[0];
            nidx[(size_t)q * KNN + kk] = best;
            nmask[(size_t)q * KNN + kk] = (rv[0] <= r2) ? 1 : 0;
            sd[best] = 3.4e38f;
        }
        __syncthreads();
    }
}

// ---------------------------------------------------------------------------
// SA gather: rows = nq*KNN, cols = ldA (padded). [x_j | pos_j - pos_i | 0...]
// ---------------------------------------------------------------------------
__global__ void sa_gather_kernel(const float* __restrict__ Xsrc, int Cx,
                                 const float* __restrict__ pos_src,
                                 const float* __restrict__ pos_q,
                                 const int* __restrict__ nidx, int q0, int nq,
                                 bf16* __restrict__ Aout, int ldA) {
    long long total = (long long)nq * KNN * ldA;
    long long stride = (long long)gridDim.x * blockDim.x;
    for (long long i = (long long)blockIdx.x * blockDim.x + threadIdx.x; i < total; i += stride) {
        int c = (int)(i % ldA);
        long long r = i / ldA;
        int k = (int)(r % KNN);
        int ql = (int)(r / KNN);
        int q = q0 + ql;
        int j = nidx[(size_t)q * KNN + k];
        float v;
        if (c < Cx)            v = Xsrc[(size_t)j * Cx + c];
        else if (c < Cx + 3) { int a = c - Cx; v = pos_src[3*j + a] - pos_q[3*q + a]; }
        else                   v = 0.0f;
        Aout[(size_t)r * ldA + c] = (bf16)v;
    }
}

// ---------------------------------------------------------------------------
// Tiled bf16 WMMA GEMM: Y[R,Co] = act(A[R,Kp] * B[Kp,Co] + bias)
// Block 256 threads = 8 waves in a 4(M) x 2(N) grid; block tile 128x128;
// each wave computes 32x64 via acc[2][4] (8 WMMAs / K-step). K stepped by 32
// through LDS. R, Co multiples of 128; Kp multiple of 32.
// ---------------------------------------------------------------------------
template <bool RELU, bool OUTF32>
__global__ __launch_bounds__(256)
void wmma_gemm_kernel(const bf16* __restrict__ A, const bf16* __restrict__ Bw,
                      const float* __restrict__ bias,
                      float* __restrict__ Yf, bf16* __restrict__ Yh,
                      int R, int Kp, int Co) {
    __shared__ bf16 As[128 * 32];     // [row][k]
    __shared__ bf16 Bst[128 * 32];    // transposed: [n][k]

    int tid  = threadIdx.x;
    int lane = tid & 31;
    int w    = tid >> 5;
    int wm   = w >> 1;                // 0..3 -> rows wm*32
    int wn   = w & 1;                 // 0..1 -> cols wn*64
    int bM = blockIdx.y * 128, bN = blockIdx.x * 128;

    v8f zero = {0.f,0.f,0.f,0.f,0.f,0.f,0.f,0.f};
    v8f acc[2][4] = {{zero, zero, zero, zero}, {zero, zero, zero, zero}};

    int m    = lane & 15;
    int hiL  = (lane >> 4) & 1;
    int kaLo = hiL ? 8 : 0;
    int kaHi = hiL ? 24 : 16;
    int kb   = hiL ? 8 : 0;

    for (int k0 = 0; k0 < Kp; k0 += 32) {
        // A tile 128x32 -> As : thread t handles row t>>1, 16-col half (t&1)
        {
            int row = tid >> 1, col = (tid & 1) * 16;
            const v8bf* src = (const v8bf*)(A + (size_t)(bM + row) * Kp + k0 + col);
            v8bf a0 = src[0], a1 = src[1];
            *(v8bf*)&As[row * 32 + col]     = a0;
            *(v8bf*)&As[row * 32 + col + 8] = a1;
        }
        // B tile 32x128 -> Bst (transposed): thread t handles k-row t>>3, col seg (t&7)*16
        {
            int row = tid >> 3, col = (tid & 7) * 16;
            const v8bf* src = (const v8bf*)(Bw + (size_t)(k0 + row) * Co + bN + col);
            v8bf b0 = src[0], b1 = src[1];
            #pragma unroll
            for (int i = 0; i < 8; ++i) Bst[(col + i) * 32 + row] = b0[i];
            #pragma unroll
            for (int i = 0; i < 8; ++i) Bst[(col + 8 + i) * 32 + row] = b1[i];
        }
        __syncthreads();

        // Fragment loads per CDNA5 wave32 WMMA layouts.
        v16bf afr[2], bfr[4];
        #pragma unroll
        for (int i = 0; i < 2; ++i) {
            int r = wm * 32 + i * 16 + m;
            v8bf lo = *(const v8bf*)&As[r * 32 + kaLo];
            v8bf hi = *(const v8bf*)&As[r * 32 + kaHi];
            afr[i] = __builtin_shufflevector(lo, hi, 0,1,2,3,4,5,6,7,8,9,10,11,12,13,14,15);
        }
        #pragma unroll
        for (int j = 0; j < 4; ++j) {
            int n = wn * 64 + j * 16 + m;
            v8bf lo = *(const v8bf*)&Bst[n * 32 + kb];
            v8bf hi = *(const v8bf*)&Bst[n * 32 + kb + 16];
            bfr[j] = __builtin_shufflevector(lo, hi, 0,1,2,3,4,5,6,7,8,9,10,11,12,13,14,15);
        }
        #pragma unroll
        for (int i = 0; i < 2; ++i)
            #pragma unroll
            for (int j = 0; j < 4; ++j)
                acc[i][j] = __builtin_amdgcn_wmma_f32_16x16x32_bf16(
                    false, afr[i], false, bfr[j], (short)0, acc[i][j], false, false);
        __syncthreads();
    }

    // Epilogue (branch-free): C layout — VGPR v: lanes 0-15 -> M=v, 16-31 -> M=v+8.
    int mAdd = hiL ? 8 : 0;
    #pragma unroll
    for (int j = 0; j < 4; ++j) {
        int col = bN + wn * 64 + j * 16 + m;
        float bv = bias[col];
        #pragma unroll
        for (int i = 0; i < 2; ++i) {
            #pragma unroll
            for (int v = 0; v < 8; ++v) {
                int row = bM + wm * 32 + i * 16 + v + mAdd;
                float val = acc[i][j][v] + bv;
                if (RELU) val = fmaxf(val, 0.0f);
                size_t o = (size_t)row * Co + col;
                if (OUTF32) Yf[o] = val;
                else        Yh[o] = (bf16)val;
            }
        }
    }
}

// ---------------------------------------------------------------------------
// Masked max aggregation over KNN neighbors: [nq*KNN, C] bf16 -> [Mq, C] f32
// ---------------------------------------------------------------------------
__global__ void max_agg_kernel(const bf16* __restrict__ Y, const unsigned char* __restrict__ nmask,
                               int q0, int nq, int C, float* __restrict__ Xout) {
    int idx = blockIdx.x * blockDim.x + threadIdx.x;
    if (idx >= nq * C) return;
    int q = idx / C, c = idx % C;
    const unsigned char* mk = nmask + (size_t)(q0 + q) * KNN;
    const bf16* y = Y + (size_t)q * KNN * C + c;
    float mv = -3.4e38f;
    for (int k = 0; k < KNN; ++k)
        if (mk[k]) { float v = (float)y[(size_t)k * C]; mv = fmaxf(mv, v); }
    Xout[(size_t)(q0 + q) * C + c] = mv;
}

// ---------------------------------------------------------------------------
// kNN interpolate (k<=3): block per output point; thread 0 finds neighbors,
// all threads do the weighted channel sum. Writes bf16 into columns [0,C).
// ---------------------------------------------------------------------------
__global__ void knn_interp_kernel(const float* __restrict__ Xsrc,
                                  const float* __restrict__ pos_src,
                                  const float* __restrict__ pos_q,
                                  int M, int C, int k,
                                  bf16* __restrict__ Aout, int ldA) {
    __shared__ int   sid[3];
    __shared__ float sw[3];
    int q = blockIdx.x, tid = threadIdx.x;
    if (tid == 0) {
        float qx = pos_q[3*q], qy = pos_q[3*q+1], qz = pos_q[3*q+2];
        float bd[3] = {3.4e38f, 3.4e38f, 3.4e38f};
        int   bi[3] = {0, 0, 0};
        for (int i = 0; i < M; ++i) {
            float dx = pos_src[3*i] - qx, dy = pos_src[3*i+1] - qy, dz = pos_src[3*i+2] - qz;
            float d = dx*dx + dy*dy + dz*dz;
            if (d < bd[0])      { bd[2]=bd[1]; bi[2]=bi[1]; bd[1]=bd[0]; bi[1]=bi[0]; bd[0]=d; bi[0]=i; }
            else if (d < bd[1]) { bd[2]=bd[1]; bi[2]=bi[1]; bd[1]=d; bi[1]=i; }
            else if (d < bd[2]) { bd[2]=d; bi[2]=i; }
        }
        float wsum = 0.0f;
        for (int j = 0; j < k; ++j) { float wv = 1.0f / fmaxf(bd[j], 1e-16f); sw[j] = wv; wsum += wv; }
        for (int j = 0; j < k; ++j) sw[j] /= wsum;
        for (int j = 0; j < 3; ++j) sid[j] = bi[j];
    }
    __syncthreads();
    for (int c = tid; c < C; c += blockDim.x) {
        float a = 0.0f;
        for (int j = 0; j < k; ++j) a += Xsrc[(size_t)sid[j] * C + c] * sw[j];
        Aout[(size_t)q * ldA + c] = (bf16)a;
    }
}

// Concatenate skip features (f32 [Ns,Cs]) into cols [Coff,ldA) with zero pad.
__global__ void concat_skip_kernel(const float* __restrict__ skip, int Cs, int Coff,
                                   bf16* __restrict__ Aout, int Ns, int ldA) {
    int span = ldA - Coff;
    long long total = (long long)Ns * span;
    long long stride = (long long)gridDim.x * blockDim.x;
    for (long long i = (long long)blockIdx.x * blockDim.x + threadIdx.x; i < total; i += stride) {
        int q = (int)(i / span), c = (int)(i % span);
        float v = (c < Cs) ? skip[(size_t)q * Cs + c] : 0.0f;
        Aout[(size_t)q * ldA + Coff + c] = (bf16)v;
    }
}

// ===========================================================================
// Host orchestration
// ===========================================================================
struct LayerMeta { int widx, bidx, Ci, Cip, Co; };

extern "C" void kernel_launch(void* const* d_in, const int* in_sizes, int n_in,
                              void* d_out, int out_size, void* d_ws, size_t ws_size,
                              hipStream_t stream) {
    (void)in_sizes; (void)n_in; (void)out_size; (void)ws_size;

    static const LayerMeta L[16] = {
        // sa1: [6,256,256,512]
        { 2, 3,   6,  32, 256}, { 4, 5, 256, 256, 256}, { 6, 7, 256, 256, 512},
        // sa2: [515,256,256,512]
        { 8, 9, 515, 544, 256}, {10,11, 256, 256, 256}, {12,13, 256, 256, 512},
        // sa3
        {14,15, 515, 544, 256}, {16,17, 256, 256, 256}, {18,19, 256, 256, 512},
        // fp3: [1024,256,256]
        {20,21,1024,1024, 256}, {22,23, 256, 256, 256},
        // fp2: [768,256,256]
        {24,25, 768, 768, 256}, {26,27, 256, 256, 256},
        // fp1: [259,256,256,128]
        {28,29, 259, 288, 256}, {30,31, 256, 256, 256}, {32,33, 256, 256, 128},
    };
    size_t woff[16]; size_t wtot = 0;
    for (int i = 0; i < 16; ++i) { woff[i] = wtot; wtot += (size_t)L[i].Cip * L[i].Co; }

    const float* pos_all = (const float*)d_in[0];
    const float* x_all   = (const float*)d_in[1];
    float* out_all       = (float*)d_out;

    // ---- workspace layout ----
    char* base = (char*)d_ws; size_t off = 0;
    auto take = [&](size_t bytes) -> char* {
        char* p = base + off; off = (off + bytes + 255) & ~(size_t)255; return p;
    };
    bf16* wpad  = (bf16*)take(wtot * sizeof(bf16));
    bf16* bufA  = (bf16*)take((size_t)8192 * 544 * sizeof(bf16));
    bf16* bufB  = (bf16*)take((size_t)8192 * 544 * sizeof(bf16));
    int*  idx1  = (int*)take(1024 * 4);
    int*  idx2  = (int*)take(256 * 4);
    int*  idx3  = (int*)take(64 * 4);
    float* pos1 = (float*)take(1024 * 3 * 4);
    float* pos2 = (float*)take(256 * 3 * 4);
    float* pos3 = (float*)take(64 * 3 * 4);
    int*  nidx1 = (int*)take((size_t)1024 * KNN * 4);
    int*  nidx2 = (int*)take((size_t)256 * KNN * 4);
    int*  nidx3 = (int*)take((size_t)64 * KNN * 4);
    unsigned char* nmask1 = (unsigned char*)take((size_t)1024 * KNN);
    unsigned char* nmask2 = (unsigned char*)take((size_t)256 * KNN);
    unsigned char* nmask3 = (unsigned char*)take((size_t)64 * KNN);
    float* x1 = (float*)take((size_t)1024 * 512 * 4);
    float* x2 = (float*)take((size_t)256 * 512 * 4);
    float* x3 = (float*)take((size_t)64 * 512 * 4);
    float* f3 = (float*)take((size_t)256 * 256 * 4);
    float* f2 = (float*)take((size_t)1024 * 256 * 4);

    // ---- convert + pad all weights once ----
    for (int i = 0; i < 16; ++i) {
        long long tot = (long long)L[i].Cip * L[i].Co;
        int nb = (int)((tot + 255) / 256); if (nb > 4096) nb = 4096;
        pad_weights_kernel<<<nb, 256, 0, stream>>>(
            (const float*)d_in[L[i].widx], wpad + woff[i], L[i].Ci, L[i].Cip, L[i].Co);
    }

    auto gemm = [&](const bf16* Aq, int li, bf16* Yh, float* Yf, int R, int relu) {
        const LayerMeta& l = L[li];
        const bf16* Wp = wpad + woff[li];
        const float* bias = (const float*)d_in[l.bidx];
        dim3 grid(l.Co / 128, R / 128);
        if (Yf) {
            if (relu) wmma_gemm_kernel<true , true ><<<grid, 256, 0, stream>>>(Aq, Wp, bias, Yf, Yh, R, l.Cip, l.Co);
            else      wmma_gemm_kernel<false, true ><<<grid, 256, 0, stream>>>(Aq, Wp, bias, Yf, Yh, R, l.Cip, l.Co);
        } else {
            if (relu) wmma_gemm_kernel<true , false><<<grid, 256, 0, stream>>>(Aq, Wp, bias, Yf, Yh, R, l.Cip, l.Co);
            else      wmma_gemm_kernel<false, false><<<grid, 256, 0, stream>>>(Aq, Wp, bias, Yf, Yh, R, l.Cip, l.Co);
        }
    };

    auto run_sa = [&](int lbase, const float* xs, int Cx,
                      const float* psrc, const float* pq,
                      const int* nidx, const unsigned char* nmask,
                      int Mq, float* Xout) {
        const int CHQ = 128;
        for (int q0 = 0; q0 < Mq; q0 += CHQ) {
            int nq = (Mq - q0 < CHQ) ? (Mq - q0) : CHQ;
            int rows = nq * KNN;                         // multiple of 128
            long long tot = (long long)rows * L[lbase].Cip;
            int nb = (int)((tot + 255) / 256); if (nb > 8192) nb = 8192;
            sa_gather_kernel<<<nb, 256, 0, stream>>>(xs, Cx, psrc, pq, nidx, q0, nq,
                                                     bufA, L[lbase].Cip);
            gemm(bufA, lbase,     bufB, nullptr, rows, 1);
            gemm(bufB, lbase + 1, bufA, nullptr, rows, 1);
            gemm(bufA, lbase + 2, bufB, nullptr, rows, 0);
            int tot2 = nq * 512;
            max_agg_kernel<<<(tot2 + 255) / 256, 256, 0, stream>>>(bufB, nmask, q0, nq, 512, Xout);
        }
    };

    auto run_level = [&](const float* psrc, int Nsrc, int M, float r,
                         int* idxb, float* posq, int* nidx, unsigned char* nmask) {
        size_t sh_fps = (size_t)Nsrc * 4 + 256 * 8;
        fps_kernel<<<1, 256, sh_fps, stream>>>(psrc, Nsrc, M, idxb);
        gather_pos_kernel<<<(M * 3 + 255) / 256, 256, 0, stream>>>(psrc, idxb, M, posq);
        size_t sh_rad = (size_t)Nsrc * 4 + 256 * 8;
        radius_topk_kernel<<<M, 256, sh_rad, stream>>>(psrc, Nsrc, posq, r * r, nidx, nmask);
    };

    for (int b = 0; b < NB; ++b) {
        const float* posb = pos_all + (size_t)b * NPT * 3;
        const float* xb   = x_all   + (size_t)b * NPT * 3;
        float* outb       = out_all + (size_t)b * NPT * 128;

        // ---- SA level 1: 4096 -> 1024, r=0.2 ----
        run_level(posb, NPT, 1024, 0.2f, idx1, pos1, nidx1, nmask1);
        run_sa(0, xb, 3, posb, pos1, nidx1, nmask1, 1024, x1);

        // ---- SA level 2: 1024 -> 256, r=0.4 ----
        run_level(pos1, 1024, 256, 0.4f, idx2, pos2, nidx2, nmask2);
        run_sa(3, x1, 512, pos1, pos2, nidx2, nmask2, 256, x2);

        // ---- SA level 3: 256 -> 64, r=0.8 ----
        run_level(pos2, 256, 64, 0.8f, idx3, pos3, nidx3, nmask3);
        run_sa(6, x2, 512, pos2, pos3, nidx3, nmask3, 64, x3);

        // ---- FP3: interp(x3, pos3 -> pos2, k=1) ++ x2 -> MLP [1024,256,256] ----
        knn_interp_kernel<<<256, 128, 0, stream>>>(x3, pos3, pos2, 64, 512, 1, bufA, 1024);
        {
            long long tot = (long long)256 * 512;
            concat_skip_kernel<<<(int)((tot + 255) / 256), 256, 0, stream>>>(x2, 512, 512, bufA, 256, 1024);
        }
        gemm(bufA, 9,  bufB, nullptr, 256, 1);
        gemm(bufB, 10, nullptr, f3, 256, 0);

        // ---- FP2: interp(f3, pos2 -> pos1, k=3) ++ x1 -> MLP [768,256,256] ----
        knn_interp_kernel<<<1024, 128, 0, stream>>>(f3, pos2, pos1, 256, 256, 3, bufA, 768);
        {
            long long tot = (long long)1024 * 512;
            concat_skip_kernel<<<(int)((tot + 255) / 256), 256, 0, stream>>>(x1, 512, 256, bufA, 1024, 768);
        }
        gemm(bufA, 11, bufB, nullptr, 1024, 1);
        gemm(bufB, 12, nullptr, f2, 1024, 0);

        // ---- FP1: interp(f2, pos1 -> pos0, k=3) ++ x0 -> MLP [259,256,256,128] ----
        knn_interp_kernel<<<NPT, 128, 0, stream>>>(f2, pos1, posb, 1024, 256, 3, bufA, 288);
        {
            long long tot = (long long)NPT * 32;
            concat_skip_kernel<<<(int)((tot + 255) / 256), 256, 0, stream>>>(xb, 3, 256, bufA, NPT, 288);
        }
        gemm(bufA, 13, bufB, nullptr, NPT, 1);
        gemm(bufB, 14, bufA, nullptr, NPT, 1);
        gemm(bufA, 15, nullptr, outb, NPT, 0);
    }
}
